// Head_72129680769338
// MI455X (gfx1250) — matrix operand order repaired and checked by reading, not compiled, not run on hardware
//
#include <hip/hip_runtime.h>
#include <hip/hip_bf16.h>

// ---------------------------------------------------------------------------
// Single-head causal attention for MI455X (gfx1250, wave32, WMMA).
//   B=8, T=2048, C=512, H=64, fp32 in/out.
// Pipeline:
//   1) proj_qkv: Q/K/V = x @ W{q,k,v}, f16 outputs in workspace,
//      v_wmma_f32_16x16x32_f16 with f32 accumulate; x tiles staged to LDS
//      with GLOBAL_LOAD_ASYNC_TO_LDS_B128 (double buffered).
//   2) flash_attn: flash attention with online softmax; K/V tiles prefetched
//      into LDS with async-to-LDS copies (ASYNCcnt) while WMMAs run.
// ---------------------------------------------------------------------------

typedef __attribute__((ext_vector_type(16))) _Float16 v16h;
typedef __attribute__((ext_vector_type(8)))  float    v8f;
typedef __attribute__((ext_vector_type(4)))  int      v4i;

#define AB 8
#define AT 2048
#define AC 512
#define AH 64
#define ABT (AB * AT)

static __device__ __forceinline__ v8f wmma_f16(v16h a, v16h b, v8f c) {
  return __builtin_amdgcn_wmma_f32_16x16x32_f16(false, a, false, b, (short)0, c,
                                                false, false);
}

// ---- CDNA5 async global->LDS copy (ASYNCcnt) ------------------------------
#if __has_builtin(__builtin_amdgcn_global_load_async_to_lds_b128)
#define ASYNC_LDS_CP 1
static __device__ __forceinline__ void async_cp16(const void* g, void* l) {
  __builtin_amdgcn_global_load_async_to_lds_b128(
      (__attribute__((address_space(1))) v4i*)g,
      (__attribute__((address_space(3))) v4i*)l, 0, 0);
}
#else
#define ASYNC_LDS_CP 0
#endif

static __device__ __forceinline__ void wait_async0() {
#if __has_builtin(__builtin_amdgcn_s_wait_asynccnt)
  __builtin_amdgcn_s_wait_asynccnt(0);
#else
  asm volatile("s_wait_asynccnt 0" ::: "memory");
#endif
}

// Stage one 16x32 f32 x-tile into LDS (128 chunks of 16B).
static __device__ __forceinline__ void stage_x(const float* __restrict__ x,
                                               int row0, int kk, float* xsb,
                                               int tid) {
#if ASYNC_LDS_CP
  if (tid < 128) {
    int r = tid >> 3, c4 = (tid & 7) * 4;
    async_cp16(x + (size_t)(row0 + r) * AC + kk + c4, xsb + r * 32 + c4);
  }
#else
  for (int i = tid; i < 512; i += 192) {
    int r = i >> 5, c = i & 31;
    xsb[r * 32 + c] = x[(size_t)(row0 + r) * AC + kk + c];
  }
#endif
}

// Stage one 32-key K tile and V tile (4KB each, [key][h] f16) into LDS.
static __device__ __forceinline__ void stage_kv(const _Float16* __restrict__ Kb,
                                                const _Float16* __restrict__ Vb,
                                                int key0, _Float16* ktb,
                                                _Float16* vtb, int tid) {
#if ASYNC_LDS_CP
  const _Float16* ks = Kb + (size_t)key0 * AH;
  const _Float16* vs = Vb + (size_t)key0 * AH;
  async_cp16(ks + tid * 8, ktb + tid * 8);
  async_cp16(ks + (tid + 128) * 8, ktb + (tid + 128) * 8);
  async_cp16(vs + tid * 8, vtb + tid * 8);
  async_cp16(vs + (tid + 128) * 8, vtb + (tid + 128) * 8);
#else
  const uint4* ks = (const uint4*)(Kb + (size_t)key0 * AH);
  const uint4* vs = (const uint4*)(Vb + (size_t)key0 * AH);
  uint4* kd = (uint4*)ktb;
  uint4* vd = (uint4*)vtb;
  kd[tid] = ks[tid];
  kd[tid + 128] = ks[tid + 128];
  vd[tid] = vs[tid];
  vd[tid + 128] = vs[tid + 128];
#endif
}

// ---------------------------------------------------------------------------
// Kernel 1: QKV projection.  grid.x = ABT/16 (one 16-row tile per block),
// block = 192 threads = 6 waves; wave w: matrix = w/2 (Q,K,V), col-half = w%2.
// ---------------------------------------------------------------------------
__global__ __launch_bounds__(192) void proj_qkv(
    const float* __restrict__ x,
    const float* __restrict__ Wq, const float* __restrict__ Wk,
    const float* __restrict__ Wv,
    _Float16* __restrict__ Qh, _Float16* __restrict__ Kh,
    _Float16* __restrict__ Vh) {
  __shared__ float xsf[2][16 * 32];  // double-buffered f32 x tile

  const int tid  = threadIdx.x;
  const int lane = tid & 31;
  const int w    = tid >> 5;   // 0..5
  const int mat  = w >> 1;     // 0=Q 1=K 2=V
  const int n0   = (w & 1) * 32;
  const int row0 = blockIdx.x * 16;

  const float* W = (mat == 0) ? Wq : (mat == 1 ? Wk : Wv);
  _Float16* Oh   = (mat == 0) ? Qh : (mat == 1 ? Kh : Vh);

  const int m    = lane & 15;
  const int koff = (lane >> 4) * 8;   // unused; A fragment also uses kg
  const int kg   = (lane >> 4) * 16;  // B-fragment K offset for hi half-wave
  const int nA   = n0 + m;
  const int nB   = n0 + 16 + m;

  v8f acc0 = {};
  v8f acc1 = {};

  stage_x(x, row0, 0, xsf[0], tid);
  wait_async0();
  __syncthreads();

  for (int kki = 0; kki < AC / 32; ++kki) {
    const int kk  = kki * 32;
    const int cur = kki & 1;
    if (kk + 32 < AC) stage_x(x, row0, kk + 32, xsf[cur ^ 1], tid);

    // A fragment (16x32 f16): K = hi*16 + v*2 + p
    const float* xsb = xsf[cur];
    v16h a;
#pragma unroll
    for (int v = 0; v < 8; ++v) {
      int kb = kg + v * 2;
      a[2 * v]     = (_Float16)xsb[m * 32 + kb];
      a[2 * v + 1] = (_Float16)xsb[m * 32 + kb + 1];
    }

    // B fragments (32x16 f16) straight from W (L2-resident, 128KB total).
    v16h b0, b1;
#pragma unroll
    for (int e = 0; e < 16; ++e) {
      size_t krow = (size_t)(kk + kg + e) * AH;
      b0[e] = (_Float16)W[krow + nA];
      b1[e] = (_Float16)W[krow + nB];
    }

    acc0 = wmma_f16(a, b0, acc0);
    acc1 = wmma_f16(a, b1, acc1);

    wait_async0();   // prefetch landed
    __syncthreads(); // all waves done with xsf[cur] before next overwrite
  }

  // C/D layout: VGPR r -> row r + 8*hi, col = n-base + (lane&15).
  const int rowadd = (lane >> 4) * 8;
#pragma unroll
  for (int r = 0; r < 8; ++r) {
    size_t row = (size_t)(row0 + r + rowadd);
    Oh[row * AH + n0 + m]      = (_Float16)acc0[r];
    Oh[row * AH + n0 + 16 + m] = (_Float16)acc1[r];
  }
}

// ---------------------------------------------------------------------------
// Kernel 2: flash attention.  grid = (T/64, B), block = 128 threads = 4 waves.
// Wave w owns query rows q0 = 64*blockIdx.x + 16*w .. +15.
// Double-buffered K/V tiles prefetched with async-to-LDS while computing.
// ---------------------------------------------------------------------------
__global__ __launch_bounds__(128) void flash_attn(
    const _Float16* __restrict__ Qh, const _Float16* __restrict__ Kh,
    const _Float16* __restrict__ Vh, float* __restrict__ out) {
  __shared__ _Float16 ktbuf[2 * 32 * 64];  // K tiles [buf][key][h]
  __shared__ _Float16 vtbuf[2 * 32 * 64];  // V tiles [buf][key][h]
  __shared__ _Float16 pb[4][16 * 32];      // per-wave P scratch [row][key]

  const int tid  = threadIdx.x;
  const int lane = tid & 31;
  const int w    = tid >> 5;
  const int b    = blockIdx.y;
  const int q0   = blockIdx.x * 64 + w * 16;

  const _Float16* Qb = Qh + (size_t)b * AT * AH;
  const _Float16* Kb = Kh + (size_t)b * AT * AH;
  const _Float16* Vb = Vh + (size_t)b * AT * AH;

  const int m      = lane & 15;
  const int koff   = (lane >> 4) * 8;
  const int kg     = (lane >> 4) * 16;
  const int ncol   = lane & 15;
  const int rowadd = (lane >> 4) * 8;

  // Q A-fragments (register-resident for the whole key loop).
  v16h aq0, aq1;
  {
    const _Float16* qr = Qb + (size_t)(q0 + m) * AH;
#pragma unroll
    for (int v = 0; v < 8; ++v) {
      int kb = kg + v * 2;
      aq0[2 * v]     = qr[kb];
      aq0[2 * v + 1] = qr[kb + 1];
      aq1[2 * v]     = qr[32 + kb];
      aq1[2 * v + 1] = qr[32 + kb + 1];
    }
  }

  v8f O0 = {}, O1 = {}, O2 = {}, O3 = {};
  float m_[8], l_[8];
#pragma unroll
  for (int r = 0; r < 8; ++r) {
    m_[r] = -1e30f;
    l_[r] = 0.0f;
  }

  const int nchunks = blockIdx.x * 2 + 2;  // keys 0 .. 64*bx+63

  stage_kv(Kb, Vb, 0, &ktbuf[0], &vtbuf[0], tid);
  wait_async0();
  __syncthreads();

  for (int j = 0; j < nchunks; ++j) {
    const int key0 = j * 32;
    const _Float16* ktb = &ktbuf[(j & 1) * (32 * 64)];
    const _Float16* vtb = &vtbuf[(j & 1) * (32 * 64)];

    // Prefetch next chunk into the other buffer (overlaps with WMMAs below).
    if (j + 1 < nchunks)
      stage_kv(Kb, Vb, key0 + 32, &ktbuf[((j + 1) & 1) * (32 * 64)],
               &vtbuf[((j + 1) & 1) * (32 * 64)], tid);

    // Waves whose rows are fully masked skip compute (wave-uniform branch).
    if (key0 <= q0 + 15) {
      // ---- S = Q K^T : two 16x16 tiles (keys key0..+15, key0+16..+31) ----
      v8f s0 = {}, s1 = {};
      v16h bk;
#pragma unroll
      for (int e = 0; e < 16; ++e) bk[e] = ktb[ncol * 64 + kg + e];
      s0 = wmma_f16(aq0, bk, s0);
#pragma unroll
      for (int e = 0; e < 16; ++e) bk[e] = ktb[ncol * 64 + 32 + kg + e];
      s0 = wmma_f16(aq1, bk, s0);
#pragma unroll
      for (int e = 0; e < 16; ++e) bk[e] = ktb[(16 + ncol) * 64 + kg + e];
      s1 = wmma_f16(aq0, bk, s1);
#pragma unroll
      for (int e = 0; e < 16; ++e) bk[e] = ktb[(16 + ncol) * 64 + 32 + kg + e];
      s1 = wmma_f16(aq1, bk, s1);

      // ---- causal mask (reference applies no 1/sqrt(H) scaling) ----
#pragma unroll
      for (int r = 0; r < 8; ++r) {
        int qrow = q0 + r + rowadd;
        if (key0 + ncol > qrow)      s0[r] = -1e30f;
        if (key0 + 16 + ncol > qrow) s1[r] = -1e30f;
      }

      // ---- online softmax; row stats per-lane arrays (row = r + 8*hi) ----
      float sc[8];
#pragma unroll
      for (int r = 0; r < 8; ++r) {
        float mx = fmaxf(s0[r], s1[r]);
#pragma unroll
        for (int off = 1; off < 16; off <<= 1)
          mx = fmaxf(mx, __shfl_xor(mx, off, 32));
        float mn = fmaxf(m_[r], mx);
        sc[r] = __expf(m_[r] - mn);
        m_[r] = mn;
        float p0 = __expf(s0[r] - mn);
        float p1 = __expf(s1[r] - mn);
        s0[r] = p0;
        s1[r] = p1;
        float rs = p0 + p1;
#pragma unroll
        for (int off = 1; off < 16; off <<= 1) rs += __shfl_xor(rs, off, 32);
        l_[r] = l_[r] * sc[r] + rs;
      }
#pragma unroll
      for (int r = 0; r < 8; ++r) {
        O0[r] *= sc[r];
        O1[r] *= sc[r];
        O2[r] *= sc[r];
        O3[r] *= sc[r];
      }

      // ---- C-layout -> A-layout for P via per-wave LDS (in-order DS) ----
      _Float16* pw = &pb[w][0];
#pragma unroll
      for (int r = 0; r < 8; ++r) {
        int row = r + rowadd;
        pw[row * 32 + ncol]      = (_Float16)s0[r];
        pw[row * 32 + 16 + ncol] = (_Float16)s1[r];
      }
      v16h ap;
#pragma unroll
      for (int v = 0; v < 8; ++v) {
        int kb = kg + v * 2;
        ap[2 * v]     = pw[m * 32 + kb];
        ap[2 * v + 1] = pw[m * 32 + kb + 1];
      }

      // ---- O += P V : B fragments from [key][h] tile (element e = key) ----
      v16h bv;
#pragma unroll
      for (int e = 0; e < 16; ++e) bv[e] = vtb[(kg + e) * 64 + ncol];
      O0 = wmma_f16(ap, bv, O0);
#pragma unroll
      for (int e = 0; e < 16; ++e) bv[e] = vtb[(kg + e) * 64 + 16 + ncol];
      O1 = wmma_f16(ap, bv, O1);
#pragma unroll
      for (int e = 0; e < 16; ++e) bv[e] = vtb[(kg + e) * 64 + 32 + ncol];
      O2 = wmma_f16(ap, bv, O2);
#pragma unroll
      for (int e = 0; e < 16; ++e) bv[e] = vtb[(kg + e) * 64 + 48 + ncol];
      O3 = wmma_f16(ap, bv, O3);
    }

    wait_async0();   // prefetched tiles landed in LDS
    __syncthreads(); // all waves done before next-buffer overwrite
  }

  // Epilogue: out = O / l, fp32.
  float* outb = out + (size_t)b * AT * AH;
#pragma unroll
  for (int r = 0; r < 8; ++r) {
    float inv = 1.0f / l_[r];
    size_t row = (size_t)(q0 + r + rowadd);
    outb[row * AH + ncol]      = O0[r] * inv;
    outb[row * AH + 16 + ncol] = O1[r] * inv;
    outb[row * AH + 32 + ncol] = O2[r] * inv;
    outb[row * AH + 48 + ncol] = O3[r] * inv;
  }
}

// ---------------------------------------------------------------------------
extern "C" void kernel_launch(void* const* d_in, const int* in_sizes, int n_in,
                              void* d_out, int out_size, void* d_ws,
                              size_t ws_size, hipStream_t stream) {
  const float* x  = (const float*)d_in[0];
  const float* Wq = (const float*)d_in[1];
  const float* Wk = (const float*)d_in[2];
  const float* Wv = (const float*)d_in[3];
  float* out = (float*)d_out;

  // Workspace: Q/K/V as f16, 2MB each (6MB total).
  _Float16* Qh = (_Float16*)d_ws;
  _Float16* Kh = Qh + (size_t)ABT * AH;
  _Float16* Vh = Kh + (size_t)ABT * AH;

  proj_qkv<<<dim3(ABT / 16), dim3(192), 0, stream>>>(x, Wq, Wk, Wv, Qh, Kh, Vh);
  flash_attn<<<dim3(AT / 64, AB), dim3(128), 0, stream>>>(Qh, Kh, Vh, out);
}
